// AttnBlock_1992864825809
// MI455X (gfx1250) — matrix operand order repaired and checked by reading, not compiled, not run on hardware
//
#include <hip/hip_runtime.h>

#define BB 4
#define CC 512
#define NN 4096
#define D3 1536

#if defined(__has_builtin)
#  if __has_builtin(__builtin_amdgcn_tensor_load_to_lds)
#    define HAVE_TDM 1
#  endif
#endif
#if defined(HAVE_TDM) && defined(__has_include)
#  if __has_include(<hip/amd_detail/amd_gfx1250_TDM.h>)
#    define TDM_6ARG 1
#  endif
#endif

typedef _Float16 v8h  __attribute__((ext_vector_type(8)));
typedef _Float16 v16h __attribute__((ext_vector_type(16)));
typedef float    v8f  __attribute__((ext_vector_type(8)));
typedef unsigned int u32x4 __attribute__((ext_vector_type(4)));
typedef int          i32x4 __attribute__((ext_vector_type(4)));
typedef int          i32x8 __attribute__((ext_vector_type(8)));

static __device__ __forceinline__ v16h cat8(v8h lo, v8h hi) {
    return __builtin_shufflevector(lo, hi, 0,1,2,3,4,5,6,7,8,9,10,11,12,13,14,15);
}

// A fragment (16x32 f16, M x K) from a row-major [m][k] LDS tile.
static __device__ __forceinline__ v16h frag_a(const _Float16* base, int stride, int lane) {
    int m  = lane & 15;
    int kb = (lane >> 4) * 8;
    const _Float16* p = base + m * stride + kb;
    v8h lo = *(const v8h*)p;          // K = kb .. kb+7
    v8h hi = *(const v8h*)(p + 16);   // K = 16+kb .. 16+kb+7
    return cat8(lo, hi);
}

// B fragment (32x16 f16, K x N) from a [n][k] LDS tile.
static __device__ __forceinline__ v16h frag_b(const _Float16* base, int stride, int lane) {
    int n  = lane & 15;
    int kb = (lane >> 4) * 8;
    const _Float16* p = base + n * stride + kb;
    v8h lo = *(const v8h*)p;
    v8h hi = *(const v8h*)(p + 16);
    return cat8(lo, hi);
}

static __device__ __forceinline__ v8f wmma16(v16h a, v16h b, v8f c) {
    return __builtin_amdgcn_wmma_f32_16x16x32_f16(false, a, false, b, (short)0, c, false, false);
}

// Flat shared pointer -> LDS byte offset (ISA 10.2: LDS aperture addr = addr[31:0]).
#define LDSOFF(p) ((unsigned)(unsigned long long)(const void*)(p))

#ifdef HAVE_TDM
// Issue a 2-D TDM tile load: global (width_el x rows, row stride row_stride_el,
// 2-byte elements) -> LDS at lds_byte_addr, padding pad_amt after every
// pad_int interval (ISA D# codes) to produce a padded LDS row stride.
static __device__ __forceinline__ void tdm_load_2d(unsigned lds_byte_addr,
                                                   const void* gptr,
                                                   unsigned width_el, unsigned rows,
                                                   unsigned row_stride_el,
                                                   unsigned pad_int_code,
                                                   unsigned pad_amt_code) {
    unsigned long long ga = (unsigned long long)gptr;
    u32x4 g0;
    g0.x = 1u;                                    // count=1, user-mode, no gather
    g0.y = lds_byte_addr;                         // D#.lds_addr
    g0.z = (unsigned)ga;                          // D#.global_addr[31:0]
    g0.w = ((unsigned)(ga >> 32) & 0x01FFFFFFu) | (2u << 30);  // addr[56:32] | type=2
    i32x8 g1;
    g1[0] = (int)((1u << 16)                      // data_size = 2 bytes
                | (1u << 20)                      // pad_enable
                | (pad_int_code << 22)            // pad_interval
                | (pad_amt_code << 25));          // pad_amount
    g1[1] = (int)((width_el & 0xFFFFu) << 16);                    // tensor_dim0[15:0] @bit48
    g1[2] = (int)(((width_el >> 16) & 0xFFFFu)                    // tensor_dim0[31:16]
                | ((rows & 0xFFFFu) << 16));                      // tensor_dim1[15:0]
    g1[3] = (int)(((rows >> 16) & 0xFFFFu)                        // tensor_dim1[31:16]
                | ((width_el & 0xFFFFu) << 16));                  // tile_dim0
    g1[4] = (int)(rows & 0xFFFFu);                                // tile_dim1 (tile_dim2=0)
    g1[5] = (int)row_stride_el;                                   // tensor_dim0_stride[31:0]
    g1[6] = 0;                                                    // stride hi / dim1_stride lo
    g1[7] = 0;
    i32x4 z4 = {0, 0, 0, 0};
#ifdef TDM_6ARG
    i32x8 z8 = {0, 0, 0, 0, 0, 0, 0, 0};
    __builtin_amdgcn_tensor_load_to_lds(g0, g1, z4, z4, z8, 0);
#else
    __builtin_amdgcn_tensor_load_to_lds(g0, g1, z4, z4, 0);
#endif
}
#endif  // HAVE_TDM

// ---------------------------------------------------------------------------
// Kernel 1: f32 -> f16 weight conversion
// ---------------------------------------------------------------------------
__global__ void cvt_f16_kernel(const float* __restrict__ s, _Float16* __restrict__ d, int n) {
    int i = blockIdx.x * blockDim.x + threadIdx.x;
    if (i < n) d[i] = (_Float16)s[i];
}

// ---------------------------------------------------------------------------
// Kernel 2: LayerNorm over channels, (b,c,h,w) f32 -> token-major f16 t[b][n][c]
// ---------------------------------------------------------------------------
__global__ __launch_bounds__(256) void ln_kernel(const float* __restrict__ x,
                                                 const float* __restrict__ gamma,
                                                 const float* __restrict__ beta,
                                                 _Float16* __restrict__ t) {
    extern __shared__ char smem[];
    float* xs = (float*)smem;                 // [512][33] padded
    __shared__ float psum[256], psq[256], mean[32], rstd[32];

    const int tid   = threadIdx.x;
    const int b     = blockIdx.x >> 7;
    const int n0    = (blockIdx.x & 127) * 32;
    const int n_idx = tid & 31;
    const int cg    = tid >> 5;

    float s = 0.f, sq = 0.f;
    for (int c = cg; c < CC; c += 8) {
        float v = x[((size_t)b * CC + c) * NN + n0 + n_idx];
        xs[c * 33 + n_idx] = v;
        s += v; sq += v * v;
    }
    psum[cg * 32 + n_idx] = s;
    psq [cg * 32 + n_idx] = sq;
    __syncthreads();

    if (tid < 32) {
        float s2 = 0.f, q2 = 0.f;
        #pragma unroll
        for (int i = 0; i < 8; ++i) { s2 += psum[i * 32 + tid]; q2 += psq[i * 32 + tid]; }
        float mu  = s2 * (1.0f / CC);
        float var = q2 * (1.0f / CC) - mu * mu;
        mean[tid] = mu;
        rstd[tid] = rsqrtf(var + 1e-5f);
    }
    __syncthreads();

    for (int idx = tid; idx < 32 * CC; idx += 256) {
        int n = idx >> 9;
        int c = idx & (CC - 1);
        float v = (xs[c * 33 + n] - mean[n]) * rstd[n] * gamma[c] + beta[c];
        t[((size_t)b * NN + n0 + n) * CC + c] = (_Float16)v;
    }
}

// ---------------------------------------------------------------------------
// Kernel 3: QKV GEMM (M=16384, N=1536, K=512). Scale folded into Q; V stored
// transposed [c][n] (B-matrix order for the PV GEMM).
// ---------------------------------------------------------------------------
__global__ __launch_bounds__(256) void qkv_gemm_kernel(const _Float16* __restrict__ t,
                                                       const _Float16* __restrict__ wq,
                                                       _Float16* __restrict__ qb,
                                                       _Float16* __restrict__ kb,
                                                       _Float16* __restrict__ vt) {
    __shared__ _Float16 As[64 * 40];
    __shared__ _Float16 Bs[128 * 40];

    const int tid  = threadIdx.x;
    const int lane = tid & 31;
    const int wave = tid >> 5;
    const int wm   = (wave & 1) * 32;
    const int wn   = (wave >> 1) * 32;
    const int M0   = blockIdx.x * 64;
    const int d0   = blockIdx.y * 128;

    v8f acc[2][2] = {};

    for (int k0 = 0; k0 < CC; k0 += 32) {
        {
            int row = tid >> 2, seg = tid & 3;
            *(v8h*)&As[row * 40 + seg * 8] =
                *(const v8h*)&t[(size_t)(M0 + row) * CC + k0 + seg * 8];
        }
        #pragma unroll
        for (int s = 0; s < 2; ++s) {
            int idx = tid + s * 256;
            int row = idx >> 2, seg = idx & 3;
            *(v8h*)&Bs[row * 40 + seg * 8] =
                *(const v8h*)&wq[(size_t)(d0 + row) * CC + k0 + seg * 8];
        }
        __syncthreads();

        v16h a0 = frag_a(&As[wm * 40],        40, lane);
        v16h a1 = frag_a(&As[(wm + 16) * 40], 40, lane);
        v16h b0 = frag_b(&Bs[wn * 40],        40, lane);
        v16h b1 = frag_b(&Bs[(wn + 16) * 40], 40, lane);
        acc[0][0] = wmma16(a0, b0, acc[0][0]);
        acc[0][1] = wmma16(a0, b1, acc[0][1]);
        acc[1][0] = wmma16(a1, b0, acc[1][0]);
        acc[1][1] = wmma16(a1, b1, acc[1][1]);
        __syncthreads();
    }

    const float qscale = 0.044194173824159216f;  // 512^-0.5
    const int half = lane >> 4, col = lane & 15;
    #pragma unroll
    for (int i = 0; i < 2; ++i)
        #pragma unroll
        for (int j = 0; j < 2; ++j)
            #pragma unroll
            for (int e = 0; e < 8; ++e) {
                int mg = M0 + wm + i * 16 + half * 8 + e;
                int d  = d0 + wn + j * 16 + col;
                int bi = mg >> 12, n = mg & (NN - 1);
                float v = acc[i][j][e];
                if (d < CC)
                    qb[((size_t)bi * NN + n) * CC + d] = (_Float16)(v * qscale);
                else if (d < 2 * CC)
                    kb[((size_t)bi * NN + n) * CC + (d - CC)] = (_Float16)v;
                else
                    vt[((size_t)bi * CC + (d - 2 * CC)) * NN + n] = (_Float16)v;
            }
}

// ---------------------------------------------------------------------------
// Kernel 4: attention, flash-style two-pass, TDM double-buffered K/V tiles.
// ---------------------------------------------------------------------------
__global__ __launch_bounds__(256) void attn_kernel(const _Float16* __restrict__ qb,
                                                   const _Float16* __restrict__ kk,
                                                   const _Float16* __restrict__ vt,
                                                   _Float16* __restrict__ ob) {
    extern __shared__ char smem[];
    _Float16* Qs  = (_Float16*)smem;           // 32 x 520
    _Float16* Ks0 = Qs  + 32 * 520;            // 32 x 520 (buffer 0)
    _Float16* Ks1 = Ks0 + 32 * 520;            // 32 x 520 (buffer 1)
    _Float16* Vs0 = Ks1 + 32 * 520;            // 512 x 40 (buffer 0)
    _Float16* Vs1 = Vs0 + 512 * 40;            // 512 x 40 (buffer 1)
    _Float16* Ps  = Vs1 + 512 * 40;            // 32 x 40
    float* Ss      = (float*)(Ps + 32 * 40);   // 32 x 36
    float* row_m   = Ss + 32 * 36;
    float* row_l   = row_m + 32;
    float* row_inv = row_l + 32;

    const int tid  = threadIdx.x;
    const int lane = tid & 31;
    const int wave = tid >> 5;
    const int b    = blockIdx.x >> 7;
    const int q0   = (blockIdx.x & 127) * 32;
    const int half = lane >> 4, col = lane & 15;

    const _Float16* kbase = kk + (size_t)b * NN * CC;
    const _Float16* vbase = vt + (size_t)b * CC * NN;

#ifdef HAVE_TDM
    if (wave == 0) {  // Q tile + first K tile via Tensor Data Mover
        tdm_load_2d(LDSOFF(Qs),  qb + ((size_t)b * NN + q0) * CC, 512, 32, 512, 7, 3);
        tdm_load_2d(LDSOFF(Ks0), kbase,                            512, 32, 512, 7, 3);
    }
#else
    {   // synchronous Q tile load
        int row = tid >> 3, l8 = tid & 7;
        const _Float16* src = &qb[((size_t)b * NN + q0 + row) * CC];
        #pragma unroll
        for (int s = 0; s < 8; ++s) {
            int seg = l8 + s * 8;
            *(v8h*)&Qs[row * 520 + seg * 8] = *(const v8h*)&src[seg * 8];
        }
    }
#endif
    if (tid < 32) { row_m[tid] = -3.0e38f; row_l[tid] = 0.0f; }

    // ---------------- pass 1: softmax statistics ----------------
    int cur = 0;
    for (int kt = 0; kt < NN; kt += 32, cur ^= 1) {
        _Float16* Kc = cur ? Ks1 : Ks0;
        _Float16* Kn = cur ? Ks0 : Ks1;
#ifdef HAVE_TDM
        if (wave == 0) {
            if (kt + 32 < NN) {
                tdm_load_2d(LDSOFF(Kn), kbase + (size_t)(kt + 32) * CC, 512, 32, 512, 7, 3);
                __builtin_amdgcn_s_wait_tensorcnt((short)1);   // oldest tile done
            } else {
                __builtin_amdgcn_s_wait_tensorcnt((short)0);
            }
        }
#else
        {
            int row = tid >> 3, l8 = tid & 7;
            const _Float16* src = &kbase[(size_t)(kt + row) * CC];
            #pragma unroll
            for (int s = 0; s < 8; ++s) {
                int seg = l8 + s * 8;
                *(v8h*)&Kc[row * 520 + seg * 8] = *(const v8h*)&src[seg * 8];
            }
        }
        (void)Kn;
#endif
        __syncthreads();

        if (wave < 4) {
            int iq = (wave >> 1) * 16, jk = (wave & 1) * 16;
            v8f s = {};
            for (int k0 = 0; k0 < CC; k0 += 32) {
                v16h a  = frag_a(&Qs[iq * 520 + k0], 520, lane);
                v16h bf = frag_b(&Kc[jk * 520 + k0], 520, lane);
                s = wmma16(a, bf, s);
            }
            #pragma unroll
            for (int e = 0; e < 8; ++e)
                Ss[(iq + half * 8 + e) * 36 + jk + col] = s[e];
        }
        __syncthreads();

        if (tid < 32) {
            float m_old = row_m[tid];
            float tmax  = m_old;
            for (int c2 = 0; c2 < 32; ++c2) tmax = fmaxf(tmax, Ss[tid * 36 + c2]);
            float l = row_l[tid] * __expf(m_old - tmax);
            for (int c2 = 0; c2 < 32; ++c2) l += __expf(Ss[tid * 36 + c2] - tmax);
            row_m[tid] = tmax; row_l[tid] = l;
        }
        __syncthreads();
    }

    if (tid < 32) row_inv[tid] = 1.0f / row_l[tid];
#ifdef HAVE_TDM
    if (wave == 0) {  // prime pass-2 double buffers
        tdm_load_2d(LDSOFF(Ks0), kbase, 512, 32,  512,  7, 3);
        tdm_load_2d(LDSOFF(Vs0), vbase, 32,  512, 4096, 3, 3);
    }
#endif
    __syncthreads();

    // ---------------- pass 2: O = softmax(S) . V ----------------
    v8f acc[2][4] = {};
    cur = 0;
    for (int kt = 0; kt < NN; kt += 32, cur ^= 1) {
        _Float16* Kc = cur ? Ks1 : Ks0;
        _Float16* Vc = cur ? Vs1 : Vs0;
        _Float16* Kn = cur ? Ks0 : Ks1;
        _Float16* Vn = cur ? Vs0 : Vs1;
#ifdef HAVE_TDM
        if (wave == 0) {
            if (kt + 32 < NN) {
                tdm_load_2d(LDSOFF(Kn), kbase + (size_t)(kt + 32) * CC, 512, 32, 512, 7, 3);
                tdm_load_2d(LDSOFF(Vn), vbase + (kt + 32),              32, 512, 4096, 3, 3);
                __builtin_amdgcn_s_wait_tensorcnt((short)2);   // current K+V done
            } else {
                __builtin_amdgcn_s_wait_tensorcnt((short)0);
            }
        }
#else
        {
            int row = tid >> 3, l8 = tid & 7;
            const _Float16* src = &kbase[(size_t)(kt + row) * CC];
            #pragma unroll
            for (int s = 0; s < 8; ++s) {
                int seg = l8 + s * 8;
                *(v8h*)&Kc[row * 520 + seg * 8] = *(const v8h*)&src[seg * 8];
            }
        }
        #pragma unroll
        for (int s = 0; s < 2; ++s) {
            int c2 = tid + s * 256;
            const _Float16* src = &vbase[(size_t)c2 * NN + kt];
            #pragma unroll
            for (int g = 0; g < 4; ++g)
                *(v8h*)&Vc[c2 * 40 + g * 8] = *(const v8h*)&src[g * 8];
        }
        (void)Kn; (void)Vn;
#endif
        __syncthreads();

        if (wave < 4) {   // recompute S, write P = exp(s-m)/l (f16)
            int iq = (wave >> 1) * 16, jk = (wave & 1) * 16;
            v8f s = {};
            for (int k0 = 0; k0 < CC; k0 += 32) {
                v16h a  = frag_a(&Qs[iq * 520 + k0], 520, lane);
                v16h bf = frag_b(&Kc[jk * 520 + k0], 520, lane);
                s = wmma16(a, bf, s);
            }
            #pragma unroll
            for (int e = 0; e < 8; ++e) {
                int m = iq + half * 8 + e;
                float p = __expf(s[e] - row_m[m]) * row_inv[m];
                Ps[m * 40 + jk + col] = (_Float16)p;
            }
        }
        __syncthreads();

        #pragma unroll
        for (int i = 0; i < 2; ++i) {
            v16h a = frag_a(&Ps[i * 16 * 40], 40, lane);
            #pragma unroll
            for (int j = 0; j < 4; ++j) {
                v16h bf = frag_b(&Vc[(wave * 64 + j * 16) * 40], 40, lane);
                acc[i][j] = wmma16(a, bf, acc[i][j]);
            }
        }
        __syncthreads();
    }

    #pragma unroll
    for (int i = 0; i < 2; ++i)
        #pragma unroll
        for (int j = 0; j < 4; ++j)
            #pragma unroll
            for (int e = 0; e < 8; ++e) {
                int qg = q0 + i * 16 + half * 8 + e;
                int cg = wave * 64 + j * 16 + col;
                ob[((size_t)b * NN + qg) * CC + cg] = (_Float16)acc[i][j][e];
            }
}

// ---------------------------------------------------------------------------
// Kernel 5: proj GEMM + residual, write fp32 (b,c,h,w)
// ---------------------------------------------------------------------------
__global__ __launch_bounds__(256) void proj_kernel(const _Float16* __restrict__ ob,
                                                   const _Float16* __restrict__ wp,
                                                   const float* __restrict__ x,
                                                   float* __restrict__ out) {
    __shared__ _Float16 As[64 * 40];
    __shared__ _Float16 Bs[128 * 40];

    const int tid  = threadIdx.x;
    const int lane = tid & 31;
    const int wave = tid >> 5;
    const int wm   = (wave & 1) * 32;
    const int wn   = (wave >> 1) * 32;
    const int M0   = blockIdx.x * 64;
    const int d0   = blockIdx.y * 128;

    v8f acc[2][2] = {};

    for (int k0 = 0; k0 < CC; k0 += 32) {
        {
            int row = tid >> 2, seg = tid & 3;
            *(v8h*)&As[row * 40 + seg * 8] =
                *(const v8h*)&ob[(size_t)(M0 + row) * CC + k0 + seg * 8];
        }
        #pragma unroll
        for (int s = 0; s < 2; ++s) {
            int idx = tid + s * 256;
            int row = idx >> 2, seg = idx & 3;
            *(v8h*)&Bs[row * 40 + seg * 8] =
                *(const v8h*)&wp[(size_t)(d0 + row) * CC + k0 + seg * 8];
        }
        __syncthreads();

        v16h a0 = frag_a(&As[wm * 40],        40, lane);
        v16h a1 = frag_a(&As[(wm + 16) * 40], 40, lane);
        v16h b0 = frag_b(&Bs[wn * 40],        40, lane);
        v16h b1 = frag_b(&Bs[(wn + 16) * 40], 40, lane);
        acc[0][0] = wmma16(a0, b0, acc[0][0]);
        acc[0][1] = wmma16(a0, b1, acc[0][1]);
        acc[1][0] = wmma16(a1, b0, acc[1][0]);
        acc[1][1] = wmma16(a1, b1, acc[1][1]);
        __syncthreads();
    }

    const int half = lane >> 4, col = lane & 15;
    #pragma unroll
    for (int i = 0; i < 2; ++i)
        #pragma unroll
        for (int j = 0; j < 2; ++j)
            #pragma unroll
            for (int e = 0; e < 8; ++e) {
                int mg = M0 + wm + i * 16 + half * 8 + e;
                int d  = d0 + wn + j * 16 + col;
                int bi = mg >> 12, n = mg & (NN - 1);
                size_t idx = ((size_t)bi * CC + d) * NN + n;
                out[idx] = x[idx] + acc[i][j][e];
            }
}

// ---------------------------------------------------------------------------
extern "C" void kernel_launch(void* const* d_in, const int* in_sizes, int n_in,
                              void* d_out, int out_size, void* d_ws, size_t ws_size,
                              hipStream_t stream) {
    const float* x     = (const float*)d_in[0];
    const float* gamma = (const float*)d_in[1];
    const float* beta  = (const float*)d_in[2];
    const float* wqkv  = (const float*)d_in[3];
    const float* wproj = (const float*)d_in[4];
    float* out = (float*)d_out;

    const size_t EL = (size_t)BB * NN * CC;
    _Float16* t   = (_Float16*)d_ws;
    _Float16* qb  = t   + EL;
    _Float16* kb  = qb  + EL;
    _Float16* vt  = kb  + EL;
    _Float16* ob  = vt  + EL;
    _Float16* wqh = ob  + EL;
    _Float16* wph = wqh + (size_t)D3 * CC;

    cvt_f16_kernel<<<(D3 * CC + 255) / 256, 256, 0, stream>>>(wqkv, wqh, D3 * CC);
    cvt_f16_kernel<<<(CC * CC + 255) / 256, 256, 0, stream>>>(wproj, wph, CC * CC);

    ln_kernel<<<BB * (NN / 32), 256, 512 * 33 * sizeof(float), stream>>>(x, gamma, beta, t);

    qkv_gemm_kernel<<<dim3((BB * NN) / 64, D3 / 128), 256, 0, stream>>>(t, wqh, qb, kb, vt);

    size_t attn_smem = (size_t)(32 * 520 + 2 * 32 * 520 + 2 * 512 * 40 + 32 * 40) * sizeof(_Float16)
                     + (size_t)(32 * 36 + 96) * sizeof(float);   // ~189 KB < 320 KB WGP LDS
    attn_kernel<<<BB * (NN / 32), 256, attn_smem, stream>>>(qb, kb, vt, ob);

    proj_kernel<<<dim3((BB * NN) / 64, CC / 128), 256, 0, stream>>>(ob, wph, x, out);
}
